// MambaBlock_85272280695136
// MI455X (gfx1250) — matrix operand order repaired and checked by reading, not compiled, not run on hardware
//
#include <hip/hip_runtime.h>
#include <hip/hip_bf16.h>
#include <math.h>

// ---------------- problem constants ----------------
#define BB      2
#define CC      256
#define HH      48
#define LL      2304          // 48*48
#define BL      4608          // BB*LL
#define DI      512           // d_inner
#define NST     16            // D_STATE
#define DTR     16            // DT_RANK
#define DTK     32            // DT_RANK zero-padded to one WMMA K-step
#define PRJ     64            // xproj output cols (48) padded to 2 N-tiles
#define LCHUNK  128
#define NCHUNK  18            // 2304 / 128

typedef __attribute__((ext_vector_type(16))) __bf16 v16bf;
typedef __attribute__((ext_vector_type(8)))  float  v8f;

union FragBF { uint4 q[2]; unsigned int u[8]; v16bf v; };
static_assert(sizeof(FragBF) == 32, "frag size");

__device__ __forceinline__ unsigned short f2bf(float f) {
    unsigned int u = __float_as_uint(f);
    u += 0x7FFFu + ((u >> 16) & 1u);          // round-to-nearest-even
    return (unsigned short)(u >> 16);
}
__device__ __forceinline__ float siluf(float x) {
    return x / (1.0f + __expf(-x));
}

// ---------------- f32 -> bf16 raw convert ----------------
__global__ void cvt_bf16_kernel(const float* __restrict__ in,
                                unsigned short* __restrict__ out, int n) {
    int i = blockIdx.x * blockDim.x + threadIdx.x;
    if (i < n) out[i] = f2bf(in[i]);
}

// ---- f32 (rows x 16) -> bf16 (rows x 32), cols 16..31 zero-padded ----
__global__ void cvt_pad16to32_kernel(const float* __restrict__ in,
                                     unsigned short* __restrict__ out, int rows) {
    int i = blockIdx.x * blockDim.x + threadIdx.x;
    if (i < rows * 32) {
        int j = i & 31, r = i >> 5;
        out[i] = (j < 16) ? f2bf(in[r * 16 + j]) : (unsigned short)0;
    }
}

// ---- xproj weights: f32 (48 x 512) -> bf16 (64 x 512), rows 48..63 zero ----
__global__ void cvt_pad_xproj_kernel(const float* __restrict__ in,
                                     unsigned short* __restrict__ out) {
    int i = blockIdx.x * blockDim.x + threadIdx.x;   // 64*512
    if (i < PRJ * DI) {
        int r = i >> 9, c = i & 511;
        out[i] = (r < 48) ? f2bf(in[r * DI + c]) : (unsigned short)0;
    }
}

// ---------------- A = -exp(A_log), both branches ----------------
__global__ void aneg_kernel(const float* __restrict__ Af,
                            const float* __restrict__ Ab,
                            float* __restrict__ An) {
    int i = blockIdx.x * blockDim.x + threadIdx.x;   // 2*512*16
    if (i < 2 * DI * NST)
        An[i] = -__expf(i < DI * NST ? Af[i] : Ab[i - DI * NST]);
}

// ---------------- LayerNorm + positional encoding -> bf16 h ----------------
__global__ __launch_bounds__(CC) void ln_pos_kernel(
    const float* __restrict__ x, const float* __restrict__ g,
    const float* __restrict__ bta, const float* __restrict__ ps,
    unsigned short* __restrict__ hbf) {
    __shared__ float s1[8], s2[8];
    int row = blockIdx.x;                 // b*L + l
    int b = row / LL, l = row % LL;
    int c = threadIdx.x;
    float v = x[((size_t)b * CC + c) * LL + l];   // transposed read
    float vs = v, v2 = v * v;
    #pragma unroll
    for (int off = 16; off >= 1; off >>= 1) {
        vs += __shfl_down(vs, off, 32);
        v2 += __shfl_down(v2, off, 32);
    }
    int lane = c & 31, wave = c >> 5;
    if (lane == 0) { s1[wave] = vs; s2[wave] = v2; }
    __syncthreads();
    float m = 0.f, m2 = 0.f;
    #pragma unroll
    for (int w = 0; w < 8; ++w) { m += s1[w]; m2 += s2[w]; }
    m *= (1.0f / CC);
    float var = m2 * (1.0f / CC) - m * m;
    float xh = (v - m) * rsqrtf(var + 1e-5f) * g[c] + bta[c];
    // sinusoidal position
    float pv;
    const float nlog = -logf(10000.0f) * (1.0f / 128.0f);
    if (c < 128) pv = __sinf((float)l * __expf(nlog * (float)c));
    else         pv = __cosf((float)l * __expf(nlog * (float)(c - 128)));
    xh += pv * ps[0];
    hbf[(size_t)row * CC + c] = f2bf(xh);
}

// ---------------- epilogues for WMMA GEMM ----------------
// EPI: 0 plain f32 store, 1 softplus(acc+bias), 2 residual + transposed store
template <int EPI>
__device__ __forceinline__ void epi_store(
    const v8f& acc, int tileM, int tileN, int N, int m16, int hi,
    float* __restrict__ Y, const float* __restrict__ bias,
    const float* __restrict__ xres, float* __restrict__ outp,
    int Lseq, int Cdim) {
    int col = tileN * 16 + m16;
    #pragma unroll
    for (int r = 0; r < 8; ++r) {
        int row = tileM * 16 + hi * 8 + r;     // C layout: VGPR r -> M=r (+8 for hi)
        float v = acc[r];
        if (EPI == 0) {
            Y[(size_t)row * N + col] = v;
        } else if (EPI == 1) {
            float t = v + bias[col];
            Y[(size_t)row * N + col] = (t > 20.f) ? t : log1pf(__expf(t));
        } else {
            int b = row / Lseq, l = row % Lseq;
            size_t oi = ((size_t)b * Cdim + col) * Lseq + l;
            outp[oi] = xres[oi] + v;
        }
    }
}

// ---------------- WMMA bf16 GEMM: Y[M,N] = A[M,KK] * W[N,KK]^T ----------------
// KK compile-time (mult of 32); M % 128 == 0; N % 32 == 0 (padded). Each wave
// computes a 16x32 strip: A fragment reused for two WMMAs, two independent
// accumulator chains. 1-deep software pipeline on the fragment loads.
// Fragment layouts follow the ISA 16-bit A 16x32 / B 32x16 tables: each lane's
// fragment is exactly two contiguous 16-byte runs -> global_load_b128.
template <int EPI, int KK>
__global__ __launch_bounds__(256) void gemm_bf16(
    const unsigned short* __restrict__ A, const unsigned short* __restrict__ W,
    int N, float* __restrict__ Y, const float* __restrict__ bias,
    const float* __restrict__ xres, float* __restrict__ outp,
    int Lseq, int Cdim) {
    int wave = threadIdx.x >> 5;
    int lane = threadIdx.x & 31;
    int m16  = lane & 15;
    int hi   = lane >> 4;
    int tileM  = blockIdx.x * 8 + wave;
    int tileN0 = blockIdx.y * 2;

    const unsigned short* arow  = A + (size_t)(tileM * 16 + m16) * KK + hi * 8;
    const unsigned short* wrow0 = W + (size_t)(tileN0 * 16 + m16) * KK + hi * 8;
    const unsigned short* wrow1 = wrow0 + (size_t)16 * KK;

    v8f acc0 = {}, acc1 = {};
    FragBF fa, fb0, fb1;
    fa.q[0]  = *(const uint4*)(arow);
    fa.q[1]  = *(const uint4*)(arow + 16);
    fb0.q[0] = *(const uint4*)(wrow0);
    fb0.q[1] = *(const uint4*)(wrow0 + 16);
    fb1.q[0] = *(const uint4*)(wrow1);
    fb1.q[1] = *(const uint4*)(wrow1 + 16);
    #pragma unroll
    for (int k0 = 0; k0 < KK; k0 += 32) {
        FragBF na, nb0, nb1;
        if (k0 + 32 < KK) {                      // compile-time after unroll
            na.q[0]  = *(const uint4*)(arow + k0 + 32);
            na.q[1]  = *(const uint4*)(arow + k0 + 48);
            nb0.q[0] = *(const uint4*)(wrow0 + k0 + 32);
            nb0.q[1] = *(const uint4*)(wrow0 + k0 + 48);
            nb1.q[0] = *(const uint4*)(wrow1 + k0 + 32);
            nb1.q[1] = *(const uint4*)(wrow1 + k0 + 48);
        }
        acc0 = __builtin_amdgcn_wmma_f32_16x16x32_bf16(
            false, fa.v, false, fb0.v, (short)0, acc0, false, false);
        acc1 = __builtin_amdgcn_wmma_f32_16x16x32_bf16(
            false, fa.v, false, fb1.v, (short)0, acc1, false, false);
        if (k0 + 32 < KK) { fa = na; fb0 = nb0; fb1 = nb1; }
    }
    epi_store<EPI>(acc0, tileM, tileN0, N, m16, hi, Y, bias, xres, outp, Lseq, Cdim);
    epi_store<EPI>(acc1, tileM, tileN0 + 1, N, m16, hi, Y, bias, xres, outp, Lseq, Cdim);
}

// ---------------- depthwise causal conv(4) + SiLU per branch ----------------
__global__ __launch_bounds__(DI) void conv_silu_kernel(
    const float* __restrict__ xz,
    const float* __restrict__ cwf, const float* __restrict__ cbf,
    const float* __restrict__ cwb, const float* __restrict__ cbb,
    float* __restrict__ xs_f, unsigned short* __restrict__ xs_b) {
    int blk = blockIdx.x;                // (s*2+b)*L + t
    int t  = blk % LL;
    int sb = blk / LL;
    int s = sb >> 1, b = sb & 1;
    int d = threadIdx.x;
    const float* cw = s ? cwb : cwf;
    const float* cb = s ? cbb : cbf;
    float acc = cb[d];
    #pragma unroll
    for (int j = 0; j < 4; ++j) {
        int tt = t - 3 + j;
        if (tt >= 0) {
            int lm = s ? (LL - 1 - tt) : tt;
            acc += cw[d * 4 + j] * xz[((size_t)b * LL + lm) * (2 * DI) + d];
        }
    }
    float y = siluf(acc);
    size_t oi = ((size_t)sb * LL + t) * DI + d;
    xs_f[oi] = y;
    xs_b[oi] = f2bf(y);
}

// ---- pack proj[:, :16] -> bf16, zero-padded to 32 cols, for dt GEMM ----
__global__ void dtin_pack_kernel(const float* __restrict__ proj,
                                 unsigned short* __restrict__ dtin) {
    int i = blockIdx.x * blockDim.x + threadIdx.x;   // 2*BL*32
    if (i < 2 * BL * DTK) {
        int j = i & 31, row = i >> 5;
        dtin[i] = (j < 16) ? f2bf(proj[(size_t)row * PRJ + j]) : (unsigned short)0;
    }
}

// ---------------- scan pass 1: per-chunk carry (Q = h_end | h0=0, S = sum dt) ---
__global__ __launch_bounds__(256) void scan_pass1(
    const float* __restrict__ dt, const float* __restrict__ xs,
    const float* __restrict__ proj, const float* __restrict__ Aneg,
    float* __restrict__ Qb, float* __restrict__ Sb) {
    __shared__ float Bsh[LCHUNK * NST];
    int chunk = blockIdx.x, sb = blockIdx.y, dg = blockIdx.z;
    int s = sb >> 1, b = sb & 1;
    int t0 = chunk * LCHUNK;
    const float* projb = proj + ((size_t)s * BL + (size_t)b * LL + t0) * PRJ;
    for (int i = threadIdx.x; i < LCHUNK * NST; i += 256) {
        int t = i >> 4, n = i & 15;
        Bsh[i] = projb[(size_t)t * PRJ + DTR + n];
    }
    __syncthreads();
    int d = dg * 256 + threadIdx.x;
    float Areg[NST], h[NST];
    #pragma unroll
    for (int n = 0; n < NST; ++n) {
        Areg[n] = Aneg[((size_t)s * DI + d) * NST + n];
        h[n] = 0.f;
    }
    float S = 0.f;
    size_t base = ((size_t)sb * LL + t0) * DI + d;
    for (int t = 0; t < LCHUNK; ++t) {
        float dtv = dt[base + (size_t)t * DI];
        float uv  = xs[base + (size_t)t * DI];
        S += dtv;
        float du = dtv * uv;
        const float* br = &Bsh[t * NST];
        #pragma unroll
        for (int n = 0; n < NST; ++n)
            h[n] = __expf(dtv * Areg[n]) * h[n] + du * br[n];
    }
    size_t qi = (((size_t)sb * NCHUNK + chunk) * DI + d) * NST;
    #pragma unroll
    for (int n = 0; n < NST; ++n) Qb[qi + n] = h[n];
    Sb[((size_t)sb * NCHUNK + chunk) * DI + d] = S;
}

// ---------------- scan pass 2: prefix over chunks -> h0 per chunk ----------------
__global__ __launch_bounds__(256) void scan_pass2(
    const float* __restrict__ Aneg, const float* __restrict__ Qb,
    const float* __restrict__ Sb, float* __restrict__ h0b) {
    int idx = blockIdx.x * blockDim.x + threadIdx.x;   // 4*512*16 = 32768
    if (idx >= 4 * DI * NST) return;
    int n = idx & 15;
    int d = (idx >> 4) & (DI - 1);
    int sb = idx >> 13;
    int s = sb >> 1;
    float A = Aneg[((size_t)s * DI + d) * NST + n];
    float h = 0.f;
    for (int c = 0; c < NCHUNK; ++c) {
        size_t qi = (((size_t)sb * NCHUNK + c) * DI + d) * NST + n;
        h0b[qi] = h;
        float S = Sb[((size_t)sb * NCHUNK + c) * DI + d];
        h = __expf(A * S) * h + Qb[qi];
    }
}

// ---------------- scan pass 3: replay chunk from h0, emit y*silu(z) ----------------
__global__ __launch_bounds__(256) void scan_pass3(
    const float* __restrict__ dt, const float* __restrict__ xs,
    const float* __restrict__ proj, const float* __restrict__ Aneg,
    const float* __restrict__ h0b,
    const float* __restrict__ Df, const float* __restrict__ Dbk,
    const float* __restrict__ xz, float* __restrict__ ybr) {
    __shared__ float Bsh[LCHUNK * NST];
    __shared__ float Csh[LCHUNK * NST];
    int chunk = blockIdx.x, sb = blockIdx.y, dg = blockIdx.z;
    int s = sb >> 1, b = sb & 1;
    int t0 = chunk * LCHUNK;
    const float* projb = proj + ((size_t)s * BL + (size_t)b * LL + t0) * PRJ;
    for (int i = threadIdx.x; i < LCHUNK * NST; i += 256) {
        int t = i >> 4, n = i & 15;
        Bsh[i] = projb[(size_t)t * PRJ + DTR + n];
        Csh[i] = projb[(size_t)t * PRJ + DTR + NST + n];
    }
    __syncthreads();
    int d = dg * 256 + threadIdx.x;
    float Areg[NST], h[NST];
    size_t qi = (((size_t)sb * NCHUNK + chunk) * DI + d) * NST;
    #pragma unroll
    for (int n = 0; n < NST; ++n) {
        Areg[n] = Aneg[((size_t)s * DI + d) * NST + n];
        h[n] = h0b[qi + n];
    }
    float Dv = s ? Dbk[d] : Df[d];
    size_t base = ((size_t)sb * LL + t0) * DI + d;
    for (int t = 0; t < LCHUNK; ++t) {
        float dtv = dt[base + (size_t)t * DI];
        float uv  = xs[base + (size_t)t * DI];
        float du = dtv * uv;
        const float* br = &Bsh[t * NST];
        const float* cr = &Csh[t * NST];
        float y = 0.f;
        #pragma unroll
        for (int n = 0; n < NST; ++n) {
            h[n] = __expf(dtv * Areg[n]) * h[n] + du * br[n];
            y += h[n] * cr[n];
        }
        int lm = s ? (LL - 1 - (t0 + t)) : (t0 + t);
        float zv = xz[((size_t)b * LL + lm) * (2 * DI) + DI + d];
        ybr[((size_t)sb * LL + lm) * DI + d] = (y + uv * Dv) * siluf(zv);
    }
}

// ---------------- combine forward + backward branch -> bf16 ----------------
__global__ void combine_y_kernel(const float* __restrict__ ybr,
                                 unsigned short* __restrict__ yc) {
    int i = blockIdx.x * blockDim.x + threadIdx.x;   // BL*512
    if (i < BL * DI)
        yc[i] = f2bf(ybr[i] + ybr[i + (size_t)BL * DI]);
}

// ---------------- launch ----------------
extern "C" void kernel_launch(void* const* d_in, const int* in_sizes, int n_in,
                              void* d_out, int out_size, void* d_ws, size_t ws_size,
                              hipStream_t stream) {
    (void)in_sizes; (void)n_in; (void)out_size; (void)ws_size;
    const float* x          = (const float*)d_in[0];
    const float* ln_g       = (const float*)d_in[1];
    const float* ln_b       = (const float*)d_in[2];
    const float* pos_scale  = (const float*)d_in[3];
    const float* in_proj_w  = (const float*)d_in[4];
    const float* out_proj_w = (const float*)d_in[5];
    const float* conv_w_f   = (const float*)d_in[6];
    const float* conv_b_f   = (const float*)d_in[7];
    const float* xproj_w_f  = (const float*)d_in[8];
    const float* dtproj_w_f = (const float*)d_in[9];
    const float* dtproj_b_f = (const float*)d_in[10];
    const float* A_log_f    = (const float*)d_in[11];
    const float* D_f        = (const float*)d_in[12];
    const float* conv_w_b   = (const float*)d_in[13];
    const float* conv_b_b   = (const float*)d_in[14];
    const float* xproj_w_b  = (const float*)d_in[15];
    const float* dtproj_w_b = (const float*)d_in[16];
    const float* dtproj_b_b = (const float*)d_in[17];
    const float* A_log_b    = (const float*)d_in[18];
    const float* D_b        = (const float*)d_in[19];
    float* out = (float*)d_out;

    // ---- carve workspace ----
    char* w = (char*)d_ws;
    auto alloc = [&](size_t bytes) -> char* {
        char* p = w; w += (bytes + 255) & ~(size_t)255; return p;
    };
    unsigned short* w_inproj  = (unsigned short*)alloc((size_t)1024 * 256 * 2);
    unsigned short* w_xproj   = (unsigned short*)alloc((size_t)2 * PRJ * DI * 2);
    unsigned short* w_dtproj  = (unsigned short*)alloc((size_t)2 * 512 * DTK * 2);
    unsigned short* w_outproj = (unsigned short*)alloc((size_t)256 * 512 * 2);
    unsigned short* h_bf      = (unsigned short*)alloc((size_t)BL * CC * 2);
    float*          xz        = (float*)alloc((size_t)BL * 1024 * 4);
    float*          xs_f      = (float*)alloc((size_t)2 * BL * DI * 4);
    unsigned short* xs_b      = (unsigned short*)alloc((size_t)2 * BL * DI * 2);
    float*          proj      = (float*)alloc((size_t)2 * BL * PRJ * 4);
    unsigned short* dtin      = (unsigned short*)alloc((size_t)2 * BL * DTK * 2);
    float*          dtbuf     = (float*)alloc((size_t)2 * BL * DI * 4);
    float*          Aneg      = (float*)alloc((size_t)2 * DI * NST * 4);
    float*          Qb        = (float*)alloc((size_t)4 * NCHUNK * DI * NST * 4);
    float*          Sb        = (float*)alloc((size_t)4 * NCHUNK * DI * 4);
    float*          h0b       = (float*)alloc((size_t)4 * NCHUNK * DI * NST * 4);
    float*          ybr       = (float*)alloc((size_t)2 * BL * DI * 4);
    unsigned short* ycomb     = (unsigned short*)alloc((size_t)BL * DI * 2);

    // ---- weight conversions to bf16 (tiny, all fit in L2) ----
    auto cvt = [&](const float* in, unsigned short* o, int n) {
        cvt_bf16_kernel<<<(n + 255) / 256, 256, 0, stream>>>(in, o, n);
    };
    cvt(in_proj_w,  w_inproj,  1024 * 256);
    cvt_pad_xproj_kernel<<<(PRJ * DI + 255) / 256, 256, 0, stream>>>(
        xproj_w_f, w_xproj);
    cvt_pad_xproj_kernel<<<(PRJ * DI + 255) / 256, 256, 0, stream>>>(
        xproj_w_b, w_xproj + PRJ * DI);
    cvt_pad16to32_kernel<<<(512 * 32 + 255) / 256, 256, 0, stream>>>(
        dtproj_w_f, w_dtproj, 512);
    cvt_pad16to32_kernel<<<(512 * 32 + 255) / 256, 256, 0, stream>>>(
        dtproj_w_b, w_dtproj + 512 * DTK, 512);
    cvt(out_proj_w, w_outproj, 256 * 512);
    aneg_kernel<<<(2 * DI * NST + 255) / 256, 256, 0, stream>>>(A_log_f, A_log_b, Aneg);

    // ---- LN + pos -> bf16 activations ----
    ln_pos_kernel<<<BL, CC, 0, stream>>>(x, ln_g, ln_b, pos_scale, h_bf);

    // ---- in_proj GEMM: xz[BL,1024], K=256 ----
    gemm_bf16<0, 256><<<dim3(BL / 128, 1024 / 32), 256, 0, stream>>>(
        h_bf, w_inproj, 1024, xz, nullptr, nullptr, nullptr, 1, 1);

    // ---- depthwise conv + SiLU, both branches (branch-time sequences) ----
    conv_silu_kernel<<<2 * BL, DI, 0, stream>>>(
        xz, conv_w_f, conv_b_f, conv_w_b, conv_b_b, xs_f, xs_b);

    // ---- xproj GEMM per branch: proj[BL,64(padded 48)], K=512 ----
    for (int s = 0; s < 2; ++s)
        gemm_bf16<0, 512><<<dim3(BL / 128, PRJ / 32), 256, 0, stream>>>(
            xs_b + (size_t)s * BL * DI, w_xproj + (size_t)s * PRJ * DI,
            PRJ, proj + (size_t)s * BL * PRJ, nullptr, nullptr, nullptr, 1, 1);

    // ---- dt GEMM per branch with softplus epilogue, K=32 (zero-padded) ----
    dtin_pack_kernel<<<(2 * BL * DTK + 255) / 256, 256, 0, stream>>>(proj, dtin);
    for (int s = 0; s < 2; ++s)
        gemm_bf16<1, DTK><<<dim3(BL / 128, DI / 32), 256, 0, stream>>>(
            dtin + (size_t)s * BL * DTK, w_dtproj + (size_t)s * 512 * DTK,
            DI, dtbuf + (size_t)s * BL * DI,
            s ? dtproj_b_b : dtproj_b_f, nullptr, nullptr, 1, 1);

    // ---- chunked selective scan: carries -> prefix -> replay ----
    dim3 sg(NCHUNK, 4, DI / 256);
    scan_pass1<<<sg, 256, 0, stream>>>(dtbuf, xs_f, proj, Aneg, Qb, Sb);
    scan_pass2<<<(4 * DI * NST) / 256, 256, 0, stream>>>(Aneg, Qb, Sb, h0b);
    scan_pass3<<<sg, 256, 0, stream>>>(dtbuf, xs_f, proj, Aneg, h0b, D_f, D_b, xz, ybr);

    // ---- combine branches -> bf16, out_proj GEMM with residual + transpose ----
    combine_y_kernel<<<(BL * DI + 255) / 256, 256, 0, stream>>>(ybr, ycomb);
    gemm_bf16<2, 512><<<dim3(BL / 128, CC / 32), 256, 0, stream>>>(
        ycomb, w_outproj, CC, nullptr, nullptr, x, out, LL, CC);
}
